// AKOrNLayer_29180007809174
// MI455X (gfx1250) — matrix operand order repaired and checked by reading, not compiled, not run on hardware
//
#include <hip/hip_runtime.h>
#include <hip/hip_bf16.h>

// ---------------- types ----------------
typedef __attribute__((ext_vector_type(16))) __bf16        v16bf;
typedef __attribute__((ext_vector_type(8)))  float         v8f;
typedef __attribute__((ext_vector_type(8)))  unsigned int  v8u;

// ---------------- problem constants ----------------
#define NB      8192
#define NIN     1024
#define NOUT    1024
#define ND      4
#define NOD     4096          // OUT*D
#define NSTEPS  8
#define NJC     32            // 1024 / 32 K-chunks
#define TBB     16            // batch rows per block
#define VROW    1032          // padded LDS row stride (bf16 elems); 1032*2 % 16 == 0

// ---------------- helpers ----------------
__device__ __forceinline__ unsigned int f2bf_rne(float f) {
    unsigned int u = __builtin_bit_cast(unsigned int, f);
    u += 0x7FFFu + ((u >> 16) & 1u);
    return u >> 16;
}
__device__ __forceinline__ unsigned int pack2bf(float a, float b) {
    return f2bf_rne(a) | (f2bf_rne(b) << 16);
}
__device__ __forceinline__ float bf2f(unsigned short h) {
    return __builtin_bit_cast(float, (unsigned int)h << 16);
}
__device__ __forceinline__ v8f zero_v8f() {
    v8f z = {0.f, 0.f, 0.f, 0.f, 0.f, 0.f, 0.f, 0.f};
    return z;
}
__device__ __forceinline__ v16bf pack_op(uint4 lo, uint4 hc) {
    v8u u;
    u[0] = lo.x; u[1] = lo.y; u[2] = lo.z; u[3] = lo.w;
    u[4] = hc.x; u[5] = hc.y; u[6] = hc.z; u[7] = hc.w;
    return __builtin_bit_cast(v16bf, u);
}
// B tile: [half(2)][lane(32)][16B] -> two coalesced b128 loads
__device__ __forceinline__ v16bf load_btile(const unsigned int* base, int lane) {
    const uint4* p = (const uint4*)base;
    return pack_op(p[lane], p[lane + 32]);
}
// A tile row: one contiguous 32B run (K = kcol..kcol+15), 16B aligned
__device__ __forceinline__ v16bf load_atile(const uint4* row4, int kcol) {
    int base = kcol >> 3;           // uint4 index within row
    return pack_op(row4[base], row4[base + 1]);
}

// ---------------- prep: K = tanh(coupling), swizzled to WMMA B layout ----------------
__global__ __launch_bounds__(256) void prep_coupling(const float* __restrict__ coupling,
                                                     unsigned int* __restrict__ Ktp) {
    int t    = blockIdx.x * 256 + threadIdx.x;   // 524288 dwords (2 MiB)
    int tile = t >> 8;                           // 64 nt * 32 jc
    int rem  = t & 255;
    int h    = rem >> 7;
    int lane = (rem >> 2) & 31;
    int dw   = rem & 3;
    int v    = h * 4 + dw;
    int nn   = lane & 15;
    int kb   = (lane >> 4) << 4;
    int nt   = tile >> 5;
    int jc   = tile & 31;
    int i    = nt * 16 + nn;                     // N dim (output oscillator)
    int j    = jc * 32 + kb + 2 * v;             // K dim (source oscillator)
    float f0 = tanhf(coupling[i * NOUT + j]);
    float f1 = tanhf(coupling[i * NOUT + j + 1]);
    Ktp[t] = pack2bf(f0, f1);
}

// ---------------- prep: W_in swizzled to WMMA B layout (bf16) ----------------
__global__ __launch_bounds__(256) void prep_win(const float* __restrict__ W,
                                                unsigned int* __restrict__ Wp) {
    int t    = blockIdx.x * 256 + threadIdx.x;   // 2097152 dwords (8 MiB)
    int tile = t >> 8;                           // 256 nt * 32 jc
    int rem  = t & 255;
    int h    = rem >> 7;
    int lane = (rem >> 2) & 31;
    int dw   = rem & 3;
    int v    = h * 4 + dw;
    int nn   = lane & 15;
    int kb   = (lane >> 4) << 4;
    int nt   = tile >> 5;
    int jc   = tile & 31;
    int od   = nt * 16 + nn;                     // N dim (OUT*D)
    int k    = jc * 32 + kb + 2 * v;             // K dim (IN)
    float f0 = W[(size_t)k * NOD + od];
    float f1 = W[(size_t)(k + 1) * NOD + od];
    Wp[t] = pack2bf(f0, f1);
}

// ---------------- prep: x -> bf16 row-major (A-operand K-pairs are contiguous) --------
__global__ __launch_bounds__(256) void prep_x(const float* __restrict__ x,
                                              unsigned int* __restrict__ Xb) {
    int t = blockIdx.x * 256 + threadIdx.x;      // 4194304 dwords (16 MiB)
    float2 f = *(const float2*)(x + (size_t)t * 2);
    Xb[t] = pack2bf(f.x, f.y);
}

// ---------------- fused AKOrN: init GEMM + 8 Kuramoto steps in LDS ----------------
// Block: 512 threads (16 waves), 16 batch rows.  State V[(b*4+d)][o] bf16 in LDS.
__global__ __launch_bounds__(512) void akorn_fused(const unsigned int* __restrict__ Xb,
                                                   const float* __restrict__ omega,
                                                   const unsigned int* __restrict__ Ktp,
                                                   const unsigned int* __restrict__ Wp,
                                                   float* __restrict__ out) {
    extern __shared__ unsigned char smem[];
    unsigned short* Vs  = (unsigned short*)smem;   // [64][VROW] bf16
    const uint4*    Vs4 = (const uint4*)smem;

    const int lane = threadIdx.x & 31;
    const int wv   = threadIdx.x >> 5;
    const int hi   = lane >> 4;          // lane half
    const int nn   = lane & 15;
    const int kb   = hi << 4;            // A-operand K-half offset
    const int b0g  = blockIdx.x * TBB;

    // ============ init: v0 = x @ W  (M=16, N=4096, K=1024) ============
    {
        v8f acc[16];
#pragma unroll
        for (int q = 0; q < 16; ++q) acc[q] = zero_v8f();

        const uint4* xrow4 = (const uint4*)(Xb + (size_t)(b0g + nn) * (NIN / 2));
        for (int jc = 0; jc < NJC; ++jc) {
            v16bf a = load_atile(xrow4, jc * 32 + kb);   // reused across 16 N-tiles
#pragma unroll
            for (int q = 0; q < 16; ++q) {
                int nt = wv * 16 + q;
                const unsigned int* bp = Wp + (((size_t)(nt * NJC + jc)) << 8);
                if (q == 0 && jc + 1 < NJC)
                    __builtin_prefetch(Wp + (((size_t)(nt * NJC + jc + 1)) << 8), 0, 1);
                v16bf b = load_btile(bp, lane);
                acc[q]  = __builtin_amdgcn_wmma_f32_16x16x32_bf16(
                    false, a, false, b, (short)0, acc[q], false, false);
            }
        }
        // epilogue: normalize over D=4 (adjacent columns -> 4-lane groups), store bf16
#pragma unroll
        for (int q = 0; q < 16; ++q) {
            int nt = wv * 16 + q;
            int od = nt * 16 + nn;
            int o  = od >> 2;
            int d  = od & 3;
#pragma unroll
            for (int r = 0; r < 8; ++r) {
                float val = acc[q][r];
                float s   = val * val;
                s += __shfl_xor(s, 1, 32);
                s += __shfl_xor(s, 2, 32);
                float sc = 1.0f / fmaxf(sqrtf(s), 1e-12f);
                val *= sc;
                int bl = r + (hi << 3);                     // local batch row 0..15
                Vs[(bl * 4 + d) * VROW + o] = (unsigned short)f2bf_rne(val);
            }
        }
    }
    __syncthreads();

    // ============ 8 Kuramoto steps (M=64, N=1024, K=1024 per step) ============
    for (int step = 0; step < NSTEPS; ++step) {
        v8f acc[16];
#pragma unroll
        for (int q = 0; q < 16; ++q) acc[q] = zero_v8f();

        for (int jc = 0; jc < NJC; ++jc) {
            v16bf A[4];
#pragma unroll
            for (int mt = 0; mt < 4; ++mt) {               // ds_load_b128 x2 per tile
                int row = mt * 16 + nn;
                A[mt] = load_atile(Vs4 + ((row * VROW) >> 3), jc * 32 + kb);
            }
#pragma unroll
            for (int q = 0; q < 4; ++q) {
                int nt = wv * 4 + q;
                const unsigned int* bp = Ktp + (((size_t)(nt * NJC + jc)) << 8);
                if (q == 0 && jc + 1 < NJC)
                    __builtin_prefetch(Ktp + (((size_t)(nt * NJC + jc + 1)) << 8), 0, 1);
                v16bf b = load_btile(bp, lane);
#pragma unroll
                for (int mt = 0; mt < 4; ++mt)
                    acc[q * 4 + mt] = __builtin_amdgcn_wmma_f32_16x16x32_bf16(
                        false, A[mt], false, b, (short)0, acc[q * 4 + mt], false, false);
            }
        }
        __syncthreads();   // everyone done reading V for this step

        const bool last = (step == NSTEPS - 1);
#pragma unroll
        for (int q = 0; q < 4; ++q) {
            int nt = wv * 4 + q;
            int o  = nt * 16 + nn;
#pragma unroll
            for (int mt = 0; mt < 4; ++mt) {
#pragma unroll
                for (int g = 0; g < 2; ++g) {              // two oscillators per lane
                    int   bl = mt * 4 + g + hi * 2;        // local batch row
                    float w4[4];
                    float n2 = 0.f;
#pragma unroll
                    for (int d = 0; d < 4; ++d) {
                        int   M    = bl * 4 + d;           // V row index
                        float a    = acc[q * 4 + mt][g * 4 + d];
                        float vold = bf2f(Vs[M * VROW + o]);
                        float w    = vold + (a + omega[o * ND + d]);   // DT = 1.0
                        w4[d] = w;
                        n2 += w * w;
                    }
                    float sc = 1.0f / fmaxf(sqrtf(n2), 1e-12f);
#pragma unroll
                    for (int d = 0; d < 4; ++d) {
                        float w = w4[d] * sc;
                        if (last)
                            __builtin_nontemporal_store(
                                w, &out[(((size_t)(b0g + bl)) * NOUT + o) * ND + d]);
                        else
                            Vs[(bl * 4 + d) * VROW + o] = (unsigned short)f2bf_rne(w);
                    }
                }
            }
        }
        __syncthreads();   // writes visible before next step's reads
    }
}

// ---------------- host entry ----------------
extern "C" void kernel_launch(void* const* d_in, const int* in_sizes, int n_in,
                              void* d_out, int out_size, void* d_ws, size_t ws_size,
                              hipStream_t stream) {
    const float* x    = (const float*)d_in[0];   // [8192,1024]
    const float* Win  = (const float*)d_in[1];   // [1024,1024,4]
    const float* omg  = (const float*)d_in[2];   // [1024,4]
    const float* coup = (const float*)d_in[3];   // [1024,1024]
    float*       out  = (float*)d_out;           // [8192,1024,4]

    unsigned int* Ktp = (unsigned int*)d_ws;                               // 2 MiB
    unsigned int* Wp  = (unsigned int*)((char*)d_ws + (2u  << 20));        // 8 MiB
    unsigned int* Xb  = (unsigned int*)((char*)d_ws + (10u << 20));        // 16 MiB

    prep_coupling<<<2048, 256, 0, stream>>>(coup, Ktp);
    prep_win<<<8192, 256, 0, stream>>>(Win, Wp);
    prep_x<<<16384, 256, 0, stream>>>(x, Xb);

    const int lds_bytes = 64 * VROW * 2;   // 132,096 B — two blocks fit 320 KB WGP LDS
    (void)hipFuncSetAttribute((const void*)akorn_fused,
                              hipFuncAttributeMaxDynamicSharedMemorySize, lds_bytes);

    akorn_fused<<<NB / TBB, 512, lds_bytes, stream>>>(Xb, omg, Ktp, Wp, out);
}
